// MultiHeadAttention_7335804142007
// MI455X (gfx1250) — matrix operand order repaired and checked by reading, not compiled, not run on hardware
//
#include <hip/hip_runtime.h>
#include <hip/hip_bf16.h>

typedef __attribute__((ext_vector_type(16))) __bf16 v16bf;
typedef __attribute__((ext_vector_type(8)))  float  v8f;

#define B_DIM   2
#define N_SEQ   2048
#define D_MODEL 1024
#define H_NUM   16
#define HD      64
#define BN      (B_DIM * N_SEQ)

__device__ __forceinline__ v8f wmma_bf16(v16bf a, v16bf b, v8f c) {
  // (neg_a, A, neg_b, B, c_mod, C, reuse_a, reuse_b)
  return __builtin_amdgcn_wmma_f32_16x16x32_bf16(false, a, false, b, (short)0, c, false, false);
}

// ---- A fragment: 16x32 bf16 (MxK). Lane holds one row; K layout:
// VGPR j: K = 2j + 16*half (pairs), 16 consecutive K per lane half.
// Compiler fuses these into two b128 loads per fragment.
__device__ __forceinline__ v16bf load_A(const __bf16* src, int ld, int row, int k0, int hf) {
  const __bf16* p = src + (size_t)row * ld + k0;
  v16bf a;
  union { unsigned u; __bf16 h[2]; } t;
#pragma unroll
  for (int j = 0; j < 8; ++j) {
    int kk = 2 * j + 16 * hf;
    t.u = *(const unsigned*)(p + kk);
    a[2 * j]     = t.h[0];
    a[2 * j + 1] = t.h[1];
  }
  return a;
}

// ---- B fragment: 32x16 bf16 (KxN) where memory holds Bt[n][k] row-major
// (operand is Bt^T). Lane holds fixed N=lane&15, 16 consecutive K.
__device__ __forceinline__ v16bf load_Bt(const __bf16* src, int ld, int nrow, int k0, int hf) {
  const uint4* p = (const uint4*)(src + (size_t)nrow * ld + k0 + 16 * hf);
  union { uint4 u[2]; v16bf v; } t;
  t.u[0] = p[0];
  t.u[1] = p[1];
  return t.v;
}

// ---- B fragment from true row-major B[k][n]: lane fixed col, K strided by ld.
__device__ __forceinline__ v16bf load_Bcol(const __bf16* src, int ld, int k0, int col, int hf) {
  const __bf16* p = src + (size_t)(k0 + 16 * hf) * ld + col;
  v16bf b;
#pragma unroll
  for (int e = 0; e < 16; ++e) b[e] = p[(size_t)e * ld];
  return b;
}

// ======================= elementwise f32 -> bf16 =======================
__global__ void cvt_bf16_kernel(const float* __restrict__ in, __bf16* __restrict__ out, int n) {
  int i = blockIdx.x * blockDim.x + threadIdx.x;
  if (i < n) out[i] = (__bf16)in[i];
}

// ---- shared register-blocked core: 32x64 output per wave (2 M x 4 N tiles),
// Y = X @ W^T over K = D_MODEL. acc[mi][ni] per ISA C layout.
__device__ __forceinline__ void gemm_core_32x64(const __bf16* __restrict__ X,
                                                const __bf16* __restrict__ W,
                                                int row0, int col0, int l15, int hf,
                                                v8f acc[2][4]) {
#pragma unroll 1
  for (int k0 = 0; k0 < D_MODEL; k0 += 32) {
    v16bf a0 = load_A(X, D_MODEL, row0 + l15,      k0, hf);
    v16bf a1 = load_A(X, D_MODEL, row0 + 16 + l15, k0, hf);
    v16bf b0 = load_Bt(W, D_MODEL, col0 + l15,      k0, hf);
    v16bf b1 = load_Bt(W, D_MODEL, col0 + 16 + l15, k0, hf);
    v16bf b2 = load_Bt(W, D_MODEL, col0 + 32 + l15, k0, hf);
    v16bf b3 = load_Bt(W, D_MODEL, col0 + 48 + l15, k0, hf);
    acc[0][0] = wmma_bf16(a0, b0, acc[0][0]);
    acc[0][1] = wmma_bf16(a0, b1, acc[0][1]);
    acc[0][2] = wmma_bf16(a0, b2, acc[0][2]);
    acc[0][3] = wmma_bf16(a0, b3, acc[0][3]);
    acc[1][0] = wmma_bf16(a1, b0, acc[1][0]);
    acc[1][1] = wmma_bf16(a1, b1, acc[1][1]);
    acc[1][2] = wmma_bf16(a1, b2, acc[1][2]);
    acc[1][3] = wmma_bf16(a1, b3, acc[1][3]);
  }
}

// ===== projection: Out[b,h,n,d] = (X @ W^T)[b,n, d*H + h]  (head INNER) =====
// grid = (BN/32, D/64), block = 32 (one wave).
__global__ void proj_kernel(const __bf16* __restrict__ X, const __bf16* __restrict__ W,
                            __bf16* __restrict__ Out) {
  int row0 = blockIdx.x * 32, col0 = blockIdx.y * 64;
  int lane = threadIdx.x & 31, hf = lane >> 4, l15 = lane & 15;
  v8f acc[2][4] = {};
  gemm_core_32x64(X, W, row0, col0, l15, hf, acc);
  // feature f = col0 + ni*16 + N -> d = f>>4 (H==16), h = l15
  int h = l15;
#pragma unroll
  for (int mi = 0; mi < 2; ++mi) {
#pragma unroll
    for (int ni = 0; ni < 4; ++ni) {
      int d = (col0 >> 4) + ni;
#pragma unroll
      for (int r = 0; r < 8; ++r) {
        int gm = row0 + mi * 16 + r + 8 * hf;     // global token row
        int bb = gm >> 11, nn = gm & (N_SEQ - 1); // N_SEQ = 2048
        Out[(((size_t)bb * H_NUM + h) * N_SEQ + nn) * HD + d] = (__bf16)acc[mi][ni][r];
      }
    }
  }
}

// ===== output projection: out = concat @ W_o^T, fp32 store =====
__global__ void gemm_out_kernel(const __bf16* __restrict__ X, const __bf16* __restrict__ W,
                                float* __restrict__ Out) {
  int row0 = blockIdx.x * 32, col0 = blockIdx.y * 64;
  int lane = threadIdx.x & 31, hf = lane >> 4, l15 = lane & 15;
  v8f acc[2][4] = {};
  gemm_core_32x64(X, W, row0, col0, l15, hf, acc);
#pragma unroll
  for (int mi = 0; mi < 2; ++mi) {
#pragma unroll
    for (int ni = 0; ni < 4; ++ni) {
#pragma unroll
      for (int r = 0; r < 8; ++r) {
        int gm = row0 + mi * 16 + r + 8 * hf;
        Out[(size_t)gm * D_MODEL + col0 + ni * 16 + l15] = acc[mi][ni][r];
      }
    }
  }
}

// ===== fused flash attention per (b, h, 32-query tile); one wave/block =====
__global__ void attn_kernel(const __bf16* __restrict__ Q, const __bf16* __restrict__ K,
                            const __bf16* __restrict__ V, __bf16* __restrict__ Concat) {
  int idx = blockIdx.x;
  int qtile = idx & (N_SEQ / 32 - 1); idx >>= 6;  // N/32 = 64
  int h = idx & (H_NUM - 1);
  int b = idx >> 4;
  const size_t head_off = ((size_t)b * H_NUM + h) * N_SEQ * HD;
  const __bf16* Qh = Q + head_off;
  const __bf16* Kh = K + head_off;
  const __bf16* Vh = V + head_off;

  int lane = threadIdx.x & 31, hf = lane >> 4, l15 = lane & 15;
  const float sm_scale = 0.125f; // 1/sqrt(64)

  // Q A-fragments (loop-invariant): 2 M-tiles x 2 K-steps
  v16bf aq[2][2];
#pragma unroll
  for (int mt = 0; mt < 2; ++mt) {
    aq[mt][0] = load_A(Qh, HD, qtile * 32 + mt * 16 + l15, 0,  hf);
    aq[mt][1] = load_A(Qh, HD, qtile * 32 + mt * 16 + l15, 32, hf);
  }

  v8f o[2][4] = {};
  float mrow[2][8], lrow[2][8];
#pragma unroll
  for (int mt = 0; mt < 2; ++mt)
#pragma unroll
    for (int r = 0; r < 8; ++r) { mrow[mt][r] = -3.0e38f; lrow[mt][r] = 0.0f; }

  __shared__ __align__(16) __bf16 Plds[32 * 32];

#pragma unroll 1
  for (int kc = 0; kc < N_SEQ; kc += 32) {
    // prefetch next K/V chunk (speculative; lowers to global_prefetch_b8)
    __builtin_prefetch(Kh + (size_t)(kc + 32) * HD, 0, 0);
    __builtin_prefetch(Vh + (size_t)(kc + 32) * HD, 0, 0);

    // K fragments shared by both M-tiles: 2 key tiles x 2 K-steps
    v16bf bk00 = load_Bt(Kh, HD, kc + l15,      0,  hf);
    v16bf bk01 = load_Bt(Kh, HD, kc + l15,      32, hf);
    v16bf bk10 = load_Bt(Kh, HD, kc + 16 + l15, 0,  hf);
    v16bf bk11 = load_Bt(Kh, HD, kc + 16 + l15, 32, hf);

    v8f s[2][2] = {};
#pragma unroll
    for (int mt = 0; mt < 2; ++mt) {
      s[mt][0] = wmma_bf16(aq[mt][0], bk00, s[mt][0]);
      s[mt][0] = wmma_bf16(aq[mt][1], bk01, s[mt][0]);
      s[mt][1] = wmma_bf16(aq[mt][0], bk10, s[mt][1]);
      s[mt][1] = wmma_bf16(aq[mt][1], bk11, s[mt][1]);
    }

    // online softmax; row r+8*hf of each M-tile lives across 16 lanes
#pragma unroll
    for (int mt = 0; mt < 2; ++mt) {
      float corr[8];
#pragma unroll
      for (int r = 0; r < 8; ++r) {
        float v0 = s[mt][0][r] * sm_scale;
        float v1 = s[mt][1][r] * sm_scale;
        float mx = fmaxf(v0, v1);
#pragma unroll
        for (int m = 1; m < 16; m <<= 1) mx = fmaxf(mx, __shfl_xor(mx, m, 32));
        float mnew = fmaxf(mrow[mt][r], mx);
        corr[r] = __expf(mrow[mt][r] - mnew);
        mrow[mt][r] = mnew;
        float p0 = __expf(v0 - mnew);
        float p1 = __expf(v1 - mnew);
        s[mt][0][r] = p0; s[mt][1][r] = p1;
        float rs = p0 + p1;
#pragma unroll
        for (int m = 1; m < 16; m <<= 1) rs += __shfl_xor(rs, m, 32);
        lrow[mt][r] = lrow[mt][r] * corr[r] + rs;
      }
#pragma unroll
      for (int r = 0; r < 8; ++r) {
        o[mt][0][r] *= corr[r]; o[mt][1][r] *= corr[r];
        o[mt][2][r] *= corr[r]; o[mt][3][r] *= corr[r];
      }
    }

    // C-layout P -> LDS -> A-layout P (32 queries x 32 keys, bf16)
    __syncthreads();
#pragma unroll
    for (int mt = 0; mt < 2; ++mt)
#pragma unroll
      for (int r = 0; r < 8; ++r) {
        int qr = mt * 16 + r + 8 * hf;
        Plds[qr * 32 + l15]      = (__bf16)s[mt][0][r];
        Plds[qr * 32 + 16 + l15] = (__bf16)s[mt][1][r];
      }
    __syncthreads();
    v16bf ap0 = load_A(Plds, 32, l15,      0, hf);
    v16bf ap1 = load_A(Plds, 32, 16 + l15, 0, hf);

    // O += P @ V ; V fragments shared by both M-tiles
#pragma unroll
    for (int t = 0; t < 4; ++t) {
      v16bf bv = load_Bcol(Vh, HD, kc, t * 16 + l15, hf);
      o[0][t] = wmma_bf16(ap0, bv, o[0][t]);
      o[1][t] = wmma_bf16(ap1, bv, o[1][t]);
    }
  }

  // normalize + store concat[b, n, h*hd + d]  (head OUTER here)
#pragma unroll
  for (int mt = 0; mt < 2; ++mt)
#pragma unroll
    for (int r = 0; r < 8; ++r) {
      int n = qtile * 32 + mt * 16 + r + 8 * hf;
      float inv_l = 1.0f / lrow[mt][r];
      size_t base = ((size_t)b * N_SEQ + n) * D_MODEL + h * HD + l15;
      Concat[base + 0]  = (__bf16)(o[mt][0][r] * inv_l);
      Concat[base + 16] = (__bf16)(o[mt][1][r] * inv_l);
      Concat[base + 32] = (__bf16)(o[mt][2][r] * inv_l);
      Concat[base + 48] = (__bf16)(o[mt][3][r] * inv_l);
    }
}

extern "C" void kernel_launch(void* const* d_in, const int* in_sizes, int n_in,
                              void* d_out, int out_size, void* d_ws, size_t ws_size,
                              hipStream_t stream) {
  const float* x  = (const float*)d_in[0];
  const float* Wq = (const float*)d_in[1];
  const float* Wk = (const float*)d_in[2];
  const float* Wv = (const float*)d_in[3];
  const float* Wo = (const float*)d_in[4];
  float* out = (float*)d_out;

  char* ws = (char*)d_ws;
  const size_t MB = 1u << 20;
  __bf16* xb  = (__bf16*)(ws + 0 * MB);   // BN*D bf16     = 8 MB
  __bf16* wqb = (__bf16*)(ws + 8 * MB);   // D*D bf16      = 2 MB
  __bf16* wkb = (__bf16*)(ws + 10 * MB);
  __bf16* wvb = (__bf16*)(ws + 12 * MB);
  __bf16* wob = (__bf16*)(ws + 14 * MB);
  __bf16* Qr  = (__bf16*)(ws + 16 * MB);  // [B,H,N,hd]    = 8 MB
  __bf16* Kr  = (__bf16*)(ws + 24 * MB);
  __bf16* Vr  = (__bf16*)(ws + 32 * MB);
  __bf16* Cc  = (__bf16*)(ws + 40 * MB);  // concat bf16   = 8 MB  (total 48 MB)

  // fp32 -> bf16 conversions
  int nx = BN * D_MODEL;
  cvt_bf16_kernel<<<nx / 256, 256, 0, stream>>>(x, xb, nx);
  int nw = D_MODEL * D_MODEL;
  cvt_bf16_kernel<<<nw / 256, 256, 0, stream>>>(Wq, wqb, nw);
  cvt_bf16_kernel<<<nw / 256, 256, 0, stream>>>(Wk, wkb, nw);
  cvt_bf16_kernel<<<nw / 256, 256, 0, stream>>>(Wv, wvb, nw);
  cvt_bf16_kernel<<<nw / 256, 256, 0, stream>>>(Wo, wob, nw);

  // Q/K/V projections with head-split scatter (32x64 tile per wave)
  dim3 g(BN / 32, D_MODEL / 64);
  proj_kernel<<<g, 32, 0, stream>>>(xb, wqb, Qr);
  proj_kernel<<<g, 32, 0, stream>>>(xb, wkb, Kr);
  proj_kernel<<<g, 32, 0, stream>>>(xb, wvb, Vr);

  // fused attention (flash-style, 32 queries per wave)
  attn_kernel<<<B_DIM * H_NUM * (N_SEQ / 32), 32, 0, stream>>>(Qr, Kr, Vr, Cc);

  // output projection
  gemm_out_kernel<<<g, 32, 0, stream>>>(Cc, wob, out);
}
